// MultiViewImpostorNetwork_1872605741925
// MI455X (gfx1250) — matrix operand order repaired and checked by reading, not compiled, not run on hardware
//
#include <hip/hip_runtime.h>
#include <hip/hip_bf16.h>

// ---------------------------------------------------------------------------
// MultiViewImpostorNetwork — CDNA5 (gfx1250, wave32) implementation.
//   conv2/conv3 : implicit-GEMM via V_WMMA_F32_16X16X32_F16 (k-step = one
//                 3x3 tap x 32 input channels), weights staged in LDS.
//   Q/K/V/O/ray : 64x64 linears via the same WMMA tile kernel.
//   conv1/gather/softmax : VALU (K too small / data-dependent).
// All intermediates are f16 channels-last so every WMMA A-fragment is two
// contiguous 16B loads and every B-fragment is one 32B LDS load.
// ---------------------------------------------------------------------------

typedef __attribute__((ext_vector_type(16))) _Float16 v16h;
typedef __attribute__((ext_vector_type(8)))  _Float16 v8h;
typedef __attribute__((ext_vector_type(4)))  _Float16 v4h;
typedef __attribute__((ext_vector_type(8)))  float    v8f;
typedef __attribute__((ext_vector_type(4)))  float    v4f;

__device__ __forceinline__ v8f wmma_f16(v16h a, v16h b, v8f c) {
  // (neg_a, A, neg_b, B, c_mod, C, reuse_a, reuse_b)
  return __builtin_amdgcn_wmma_f32_16x16x32_f16(false, a, false, b, (short)0, c,
                                                false, false);
}

__device__ __forceinline__ v16h join8(v8h lo, v8h hi) {
  v16h r;
#pragma unroll
  for (int i = 0; i < 8; ++i) { r[i] = lo[i]; r[i + 8] = hi[i]; }
  return r;
}

__device__ __forceinline__ float leaky(float x) {
  return x >= 0.0f ? x : 0.01f * x;
}

// A-fragment for 16x16x32 f16 WMMA: per-lane k offsets are
//   regs 0..7 -> k = half*16 + 0..15 (contiguous, matches B-frag)
__device__ __forceinline__ v16h load_afrag(const _Float16* __restrict__ p, int half) {
  return join8(*(const v8h*)(p + half * 16), *(const v8h*)(p + half * 16 + 8));
}

// ---------------------------------------------------------------------------
// Weight packing: f32 -> f16, layout Wp[(ktile*64 + n)*32 + kc] so a B-frag
// (16 contiguous k values for column n) is one aligned 32-byte load.
// ---------------------------------------------------------------------------

__global__ void __launch_bounds__(256)
pack64_kernel(const float* __restrict__ W, _Float16* __restrict__ Wp) {
  int i = blockIdx.x * 256 + threadIdx.x;      // 64*64, i = n*64 + k
  int n = i >> 6, k = i & 63;
  Wp[(size_t)(k >> 5) * 2048 + n * 32 + (k & 31)] = (_Float16)W[i];
}

__global__ void __launch_bounds__(256)
packc2_kernel(const float* __restrict__ W, _Float16* __restrict__ Wp) {
  int i = blockIdx.x * 256 + threadIdx.x;      // 64*32*9, i = (n*32+cin)*9 + t
  int t = i % 9; int nc = i / 9; int n = nc >> 5; int cin = nc & 31;
  Wp[(size_t)(t * 64 + n) * 32 + cin] = (_Float16)W[i];
}

__global__ void __launch_bounds__(256)
packc3_kernel(const float* __restrict__ W, _Float16* __restrict__ Wp) {
  int i = blockIdx.x * 256 + threadIdx.x;      // 64*64*9, i = (n*64+cin)*9 + pos
  int pos = i % 9; int nc = i / 9; int n = nc >> 6; int cin = nc & 63;
  int t = pos * 2 + (cin >> 5);
  Wp[(size_t)(t * 64 + n) * 32 + (cin & 31)] = (_Float16)W[i];
}

// ---------------------------------------------------------------------------
// conv1: 3->32, 3x3, stride 1, edge pad. K=27 is too small for WMMA; direct
// VALU conv, output channels-last f16 (ready for conv2 implicit GEMM).
// ---------------------------------------------------------------------------
__global__ void __launch_bounds__(256)
conv1_kernel(const float* __restrict__ xin, const float* __restrict__ w1,
             const float* __restrict__ b1, _Float16* __restrict__ x1) {
  __shared__ float sw[864];                    // (32co,3ci,3,3)
  __shared__ float sb[32];
  for (int i = threadIdx.x; i < 864; i += 256) sw[i] = w1[i];
  if (threadIdx.x < 32) sb[threadIdx.x] = b1[threadIdx.x];
  __syncthreads();

  int id = blockIdx.x * 256 + threadIdx.x;     // over V*512*512
  int v = id >> 18, y = (id >> 9) & 511, x = id & 511;

  float acc[32];
#pragma unroll
  for (int co = 0; co < 32; ++co) acc[co] = sb[co];

#pragma unroll
  for (int p = 0; p < 9; ++p) {
    int iy = min(max(y - 1 + p / 3, 0), 511);
    int ix = min(max(x - 1 + p % 3, 0), 511);
#pragma unroll
    for (int c = 0; c < 3; ++c) {
      float xv = xin[(((size_t)(v * 3 + c)) << 18) + (iy << 9) + ix];
#pragma unroll
      for (int co = 0; co < 32; ++co)
        acc[co] = fmaf(xv, sw[(co * 3 + c) * 9 + p], acc[co]);
    }
  }
  _Float16* op = x1 + (size_t)id * 32;
#pragma unroll
  for (int co = 0; co < 32; ++co) op[co] = (_Float16)leaky(acc[co]);
}

// ---------------------------------------------------------------------------
// conv2: 32->64, 3x3, stride 2, edge pad — implicit GEMM.
// M = V*256*256 pixels, K = 9 taps x 32ch (one tap per k=32 step), N = 64.
// 8 waves/block, each wave: one 16-pixel tile x all 4 cout tiles.
// ---------------------------------------------------------------------------
__global__ void __launch_bounds__(256)
conv2_kernel(const _Float16* __restrict__ x1, const _Float16* __restrict__ Wp,
             const float* __restrict__ bias, _Float16* __restrict__ x2) {
  __shared__ __align__(32) _Float16 sW[9 * 2048];        // 36 KB
  {
    const float4* s = (const float4*)Wp; float4* d = (float4*)sW;
    for (int i = threadIdx.x; i < 9 * 256; i += 256) d[i] = s[i];
  }
  __syncthreads();

  int wave = threadIdx.x >> 5, lane = threadIdx.x & 31;
  int half = lane >> 4, lm = lane & 15;
  int row0 = blockIdx.x * 128 + wave * 16;
  int m = row0 + lm;
  int v = m >> 16, rem = m & 65535, oy = rem >> 8, ox = rem & 255;

  v8f acc[4] = {};
#pragma unroll
  for (int t = 0; t < 9; ++t) {
    int dy = t / 3, dx = t % 3;
    int iy = min(max(oy * 2 - 1 + dy, 0), 511);
    int ix = min(max(ox * 2 - 1 + dx, 0), 511);
    const _Float16* p = x1 + (((size_t)(v << 18) + (iy << 9) + ix) << 5);
    v16h a = load_afrag(p, half);
#pragma unroll
    for (int j = 0; j < 4; ++j) {
      v16h b = *(const v16h*)(sW + (size_t)t * 2048 + (j * 16 + lm) * 32 + half * 16);
      acc[j] = wmma_f16(a, b, acc[j]);
    }
  }
#pragma unroll
  for (int j = 0; j < 4; ++j) {
    int n = j * 16 + lm;
    float bn = bias[n];
#pragma unroll
    for (int r = 0; r < 8; ++r) {
      size_t mo = (size_t)(row0 + r + 8 * half);
      x2[mo * 64 + n] = (_Float16)leaky(acc[j][r] + bn);
    }
  }
}

// ---------------------------------------------------------------------------
// conv3: 64->64, 3x3, stride 2, edge pad — implicit GEMM, K = 18 k-steps
// (9 taps x 2 halves of 64 ch). Output f32 channels-last (feeds gather).
// ---------------------------------------------------------------------------
__global__ void __launch_bounds__(256)
conv3_kernel(const _Float16* __restrict__ x2, const _Float16* __restrict__ Wp,
             const float* __restrict__ bias, float* __restrict__ nts) {
  __shared__ __align__(32) _Float16 sW[18 * 2048];       // 72 KB
  {
    const float4* s = (const float4*)Wp; float4* d = (float4*)sW;
    for (int i = threadIdx.x; i < 18 * 256; i += 256) d[i] = s[i];
  }
  __syncthreads();

  int wave = threadIdx.x >> 5, lane = threadIdx.x & 31;
  int half = lane >> 4, lm = lane & 15;
  int row0 = blockIdx.x * 128 + wave * 16;
  int m = row0 + lm;
  int v = m >> 14, rem = m & 16383, oy = rem >> 7, ox = rem & 127;

  v8f acc[4] = {};
#pragma unroll
  for (int t = 0; t < 18; ++t) {
    int pos = t >> 1, cb = (t & 1) * 32;
    int dy = pos / 3, dx = pos % 3;
    int iy = min(max(oy * 2 - 1 + dy, 0), 255);
    int ix = min(max(ox * 2 - 1 + dx, 0), 255);
    const _Float16* p = x2 + (((size_t)(v << 16) + (iy << 8) + ix) << 6) + cb;
    v16h a = load_afrag(p, half);
#pragma unroll
    for (int j = 0; j < 4; ++j) {
      v16h b = *(const v16h*)(sW + (size_t)t * 2048 + (j * 16 + lm) * 32 + half * 16);
      acc[j] = wmma_f16(a, b, acc[j]);
    }
  }
#pragma unroll
  for (int j = 0; j < 4; ++j) {
    int n = j * 16 + lm;
    float bn = bias[n];
#pragma unroll
    for (int r = 0; r < 8; ++r) {
      size_t mo = (size_t)(row0 + r + 8 * half);
      nts[mo * 64 + n] = leaky(acc[j][r] + bn);
    }
  }
}

// ---------------------------------------------------------------------------
// Bilinear gather: one thread per (sample, 4-channel group). nts (32 MB)
// stays L2-resident. Writes feats f32 (output #2) and f16 (for K/V GEMMs).
// ---------------------------------------------------------------------------
__global__ void __launch_bounds__(256)
gather_kernel(const float* __restrict__ nts, const float* __restrict__ uvi,
              float* __restrict__ feats_f32, _Float16* __restrict__ feats_f16) {
  int i = blockIdx.x * 256 + threadIdx.x;      // S*16
  int s = i >> 4, c4 = i & 15;
  float u  = uvi[s * 3 + 0];
  float vv = uvi[s * 3 + 1];
  int vid  = (int)uvi[s * 3 + 2];
  float x = u * 127.0f, y = vv * 127.0f;
  int x0 = min(max((int)floorf(x), 0), 126);
  int y0 = min(max((int)floorf(y), 0), 126);
  float wx = x - (float)x0, wy = y - (float)y0;

  const float* base = nts + ((((size_t)vid * 128 + y0) * 128 + x0) << 6) + c4 * 4;
  v4f f00 = *(const v4f*)(base);
  v4f f01 = *(const v4f*)(base + 64);
  v4f f10 = *(const v4f*)(base + 128 * 64);
  v4f f11 = *(const v4f*)(base + 128 * 64 + 64);
  v4f r = f00 * ((1.0f - wx) * (1.0f - wy)) + f01 * (wx * (1.0f - wy)) +
          f10 * ((1.0f - wx) * wy)          + f11 * (wx * wy);

  *(v4f*)(feats_f32 + (size_t)s * 64 + c4 * 4) = r;
  v4h h;
#pragma unroll
  for (int c = 0; c < 4; ++c) h[c] = (_Float16)r[c];
  *(v4h*)(feats_f16 + (size_t)s * 64 + c4 * 4) = h;
}

// ---------------------------------------------------------------------------
// Ray MLP layer 1: (B,3) -> relu -> (B,64) f16. K=3 => VALU.
// ---------------------------------------------------------------------------
__global__ void __launch_bounds__(256)
rayhidden_kernel(const float* __restrict__ raydir, const float* __restrict__ rw1,
                 const float* __restrict__ rb1, _Float16* __restrict__ hidden) {
  __shared__ float sw[192];
  __shared__ float sb[64];
  for (int i = threadIdx.x; i < 192; i += 256) sw[i] = rw1[i];
  if (threadIdx.x < 64) sb[threadIdx.x] = rb1[threadIdx.x];
  __syncthreads();

  int b = blockIdx.x * 256 + threadIdx.x;
  float r0 = raydir[(size_t)b * 3], r1 = raydir[(size_t)b * 3 + 1],
        r2 = raydir[(size_t)b * 3 + 2];
  _Float16* op = hidden + (size_t)b * 64;
#pragma unroll
  for (int j = 0; j < 64; ++j) {
    float h = fmaf(r2, sw[j * 3 + 2], fmaf(r1, sw[j * 3 + 1], fmaf(r0, sw[j * 3], sb[j])));
    op[j] = (_Float16)fmaxf(h, 0.0f);
  }
}

// ---------------------------------------------------------------------------
// Generic 64-wide linear: out(M,64) = A(M,64) @ W^T + bias, f16 in, WMMA.
// 8 waves/block x 16 rows/wave = 128 rows/block (M is a multiple of 128).
// ---------------------------------------------------------------------------
template <bool OUT_F32>
__global__ void __launch_bounds__(256)
linear64_kernel(const _Float16* __restrict__ A, const _Float16* __restrict__ Wp,
                const float* __restrict__ bias, void* __restrict__ outp) {
  __shared__ __align__(32) _Float16 sW[4096];            // 2 ktiles * 64 * 32
  {
    const float4* s = (const float4*)Wp; float4* d = (float4*)sW;
    for (int i = threadIdx.x; i < 512; i += 256) d[i] = s[i];
  }
  __syncthreads();

  int wave = threadIdx.x >> 5, lane = threadIdx.x & 31;
  int half = lane >> 4, lm = lane & 15;
  size_t row0 = (size_t)blockIdx.x * 128 + wave * 16;
  const _Float16* rp = A + (row0 + lm) * 64;
  v16h a0 = load_afrag(rp, half);
  v16h a1 = load_afrag(rp + 32, half);

#pragma unroll
  for (int j = 0; j < 4; ++j) {
    int n = j * 16 + lm;
    v16h b0 = *(const v16h*)(sW + n * 32 + half * 16);
    v16h b1 = *(const v16h*)(sW + 2048 + n * 32 + half * 16);
    v8f acc = {};
    acc = wmma_f16(a0, b0, acc);
    acc = wmma_f16(a1, b1, acc);
    float bn = bias[n];
#pragma unroll
    for (int r = 0; r < 8; ++r) {
      size_t mo = row0 + r + 8 * half;
      float vo = acc[r] + bn;
      if (OUT_F32) ((float*)outp)[mo * 64 + n] = vo;
      else         ((_Float16*)outp)[mo * 64 + n] = (_Float16)vo;
    }
  }
}

// ---------------------------------------------------------------------------
// Attention core: S=3 keys, 4 heads, head_dim=16. One thread per (b, head).
// ---------------------------------------------------------------------------
__global__ void __launch_bounds__(256)
attn_kernel(const _Float16* __restrict__ q, const _Float16* __restrict__ k,
            const _Float16* __restrict__ v, _Float16* __restrict__ out) {
  int i = blockIdx.x * 256 + threadIdx.x;      // B*4
  int b = i >> 2, h = i & 3;

  const _Float16* qp = q + (size_t)b * 64 + h * 16;
  float qf[16];
#pragma unroll
  for (int d = 0; d < 16; ++d) qf[d] = (float)qp[d];

  const _Float16* kp = k + (size_t)b * 192 + h * 16;
  float s[3];
#pragma unroll
  for (int j = 0; j < 3; ++j) {
    float acc = 0.0f;
#pragma unroll
    for (int d = 0; d < 16; ++d) acc = fmaf(qf[d], (float)kp[j * 64 + d], acc);
    s[j] = acc * 0.25f;                        // 1/sqrt(16)
  }
  float mx = fmaxf(s[0], fmaxf(s[1], s[2]));
  float e0 = __expf(s[0] - mx), e1 = __expf(s[1] - mx), e2 = __expf(s[2] - mx);
  float inv = 1.0f / (e0 + e1 + e2);
  e0 *= inv; e1 *= inv; e2 *= inv;

  const _Float16* vp = v + (size_t)b * 192 + h * 16;
  _Float16* op = out + (size_t)b * 64 + h * 16;
#pragma unroll
  for (int d = 0; d < 16; ++d) {
    float o = e0 * (float)vp[d] + e1 * (float)vp[64 + d] + e2 * (float)vp[128 + d];
    op[d] = (_Float16)o;
  }
}

// ---------------------------------------------------------------------------
// Host-side orchestration.
// ---------------------------------------------------------------------------
extern "C" void kernel_launch(void* const* d_in, const int* in_sizes, int n_in,
                              void* d_out, int out_size, void* d_ws, size_t ws_size,
                              hipStream_t stream) {
  (void)n_in; (void)out_size; (void)ws_size;

  const float* view_tex = (const float*)d_in[0];
  const float* uvi      = (const float*)d_in[1];
  const float* raydir   = (const float*)d_in[2];
  const float* w1 = (const float*)d_in[3];  const float* b1 = (const float*)d_in[4];
  const float* w2 = (const float*)d_in[5];  const float* b2 = (const float*)d_in[6];
  const float* w3 = (const float*)d_in[7];  const float* b3 = (const float*)d_in[8];
  const float* rw1 = (const float*)d_in[9];  const float* rb1 = (const float*)d_in[10];
  const float* rw2 = (const float*)d_in[11]; const float* rb2 = (const float*)d_in[12];
  const float* wq = (const float*)d_in[13]; const float* bq = (const float*)d_in[14];
  const float* wk = (const float*)d_in[15]; const float* bk = (const float*)d_in[16];
  const float* wv = (const float*)d_in[17]; const float* bv = (const float*)d_in[18];
  const float* wo = (const float*)d_in[19]; const float* bo = (const float*)d_in[20];

  const int V = in_sizes[0] / (3 * 512 * 512);     // 8
  const long B = in_sizes[2] / 3;                  // 262144
  const long S = 3 * B;

  char* ws = (char*)d_ws;
  size_t off = 0;
  auto alloc = [&](size_t bytes) -> char* {
    char* p = ws + off;
    off = (off + bytes + 255) & ~(size_t)255;
    return p;
  };

  _Float16* x1      = (_Float16*)alloc((size_t)V * 512 * 512 * 32 * 2);
  _Float16* x2      = (_Float16*)alloc((size_t)V * 256 * 256 * 64 * 2);
  float*    nts     = (float*)   alloc((size_t)V * 128 * 128 * 64 * 4);
  _Float16* feats16 = (_Float16*)alloc((size_t)S * 64 * 2);
  _Float16* kproj   = (_Float16*)alloc((size_t)S * 64 * 2);
  _Float16* vproj   = (_Float16*)alloc((size_t)S * 64 * 2);
  _Float16* hidden  = (_Float16*)alloc((size_t)B * 64 * 2);
  _Float16* rayfeat = (_Float16*)alloc((size_t)B * 64 * 2);
  _Float16* qproj   = (_Float16*)alloc((size_t)B * 64 * 2);
  _Float16* attnout = (_Float16*)alloc((size_t)B * 64 * 2);
  _Float16* wp_rw2  = (_Float16*)alloc(4096 * 2);
  _Float16* wp_wq   = (_Float16*)alloc(4096 * 2);
  _Float16* wp_wk   = (_Float16*)alloc(4096 * 2);
  _Float16* wp_wv   = (_Float16*)alloc(4096 * 2);
  _Float16* wp_wo   = (_Float16*)alloc(4096 * 2);
  _Float16* wp_c2   = (_Float16*)alloc(18432 * 2);
  _Float16* wp_c3   = (_Float16*)alloc(36864 * 2);

  float* fused = (float*)d_out;                    // (B, 64)
  float* feats = fused + (size_t)B * 64;           // (B, 3, 64)

  // --- weight packing (independent, tiny) ---
  pack64_kernel<<<16, 256, 0, stream>>>(rw2, wp_rw2);
  pack64_kernel<<<16, 256, 0, stream>>>(wq, wp_wq);
  pack64_kernel<<<16, 256, 0, stream>>>(wk, wp_wk);
  pack64_kernel<<<16, 256, 0, stream>>>(wv, wp_wv);
  pack64_kernel<<<16, 256, 0, stream>>>(wo, wp_wo);
  packc2_kernel<<<72, 256, 0, stream>>>(w2, wp_c2);
  packc3_kernel<<<144, 256, 0, stream>>>(w3, wp_c3);

  // --- view encoder ---
  conv1_kernel<<<V * 512 * 512 / 256, 256, 0, stream>>>(view_tex, w1, b1, x1);
  conv2_kernel<<<V * 256 * 256 / 128, 256, 0, stream>>>(x1, wp_c2, b2, x2);
  conv3_kernel<<<V * 128 * 128 / 128, 256, 0, stream>>>(x2, wp_c3, b3, nts);

  // --- gather (writes output #2 and f16 copy for K/V GEMMs) ---
  gather_kernel<<<(int)(S * 16 / 256), 256, 0, stream>>>(nts, uvi, feats, feats16);

  // --- ray MLP ---
  rayhidden_kernel<<<(int)(B / 256), 256, 0, stream>>>(raydir, rw1, rb1, hidden);
  linear64_kernel<false><<<(int)(B / 128), 256, 0, stream>>>(hidden, wp_rw2, rb2, rayfeat);

  // --- MHA projections (WMMA GEMMs) ---
  linear64_kernel<false><<<(int)(B / 128), 256, 0, stream>>>(rayfeat, wp_wq, bq, qproj);
  linear64_kernel<false><<<(int)(S / 128), 256, 0, stream>>>(feats16, wp_wk, bk, kproj);
  linear64_kernel<false><<<(int)(S / 128), 256, 0, stream>>>(feats16, wp_wv, bv, vproj);

  // --- attention core + output projection (writes output #1) ---
  attn_kernel<<<(int)(B * 4 / 256), 256, 0, stream>>>(qproj, kproj, vproj, attnout);
  linear64_kernel<true><<<(int)(B / 128), 256, 0, stream>>>(attnout, wp_wo, bo, fused);
}
